// SonataCFMForExport_43576738185945
// MI455X (gfx1250) — matrix operand order repaired and checked by reading, not compiled, not run on hardware
//
#include <hip/hip_runtime.h>
#include <hip/hip_bf16.h>

// ---------------------------------------------------------------------------
// CDNA5 (gfx1250) wave32 WMMA implementation of the SonataCFM forward pass.
// All large matmuls use v_wmma_f32_16x16x32_bf16 (fp32 accumulate, bf16 in).
// bf16 conversion is amortized (every tensor/weight converted exactly once);
// GEMM is software-pipelined (double-buffered fragments) so loads for step
// i+1 overlap the WMMAs of step i.
// ---------------------------------------------------------------------------

#define L_LAYERS 12
#define DMODEL   512
#define NHEAD    8
#define HEADD    64
#define FFN      2048
#define TEMB     256
#define SPK      192
#define MEL      80
#define BATCH    4
#define TLEN     1024
#define NROWS    (BATCH * TLEN)      // 4096
#define EPS      1e-5f

typedef __attribute__((ext_vector_type(16))) __bf16 v16bf;
typedef __attribute__((ext_vector_type(8)))  float  v8f;
typedef unsigned short u16;

union AFrag { v16bf v; u16 u[16]; uint4 q2[2]; };
union CFrag { v8f   v; float f[8]; };

// float -> bf16 bits, round-to-nearest-even
static __device__ __forceinline__ u16 f2bf(float f) {
  union { float f; unsigned u; } c; c.f = f;
  unsigned u = c.u + 0x7FFFu + ((c.u >> 16) & 1u);
  return (u16)(u >> 16);
}
static __device__ __forceinline__ float bf2f(u16 h) {
  union { unsigned u; float f; } c; c.u = ((unsigned)h) << 16;
  return c.f;
}

static __device__ __forceinline__ v8f wmma_bf16(v16bf a, v16bf b, v8f c) {
  return __builtin_amdgcn_wmma_f32_16x16x32_bf16(false, a, false, b,
                                                 (short)0, c, false, false);
}

// ---------------------------------------------------------------------------
// fp32 -> bf16 bulk convert (vectorized, 4 elems/thread)
// ---------------------------------------------------------------------------
__global__ void k_cvt(const float* __restrict__ src, u16* __restrict__ dst,
                      int n4) {
  const int i = blockIdx.x * 256 + threadIdx.x;
  if (i < n4) {
    const float4 f = ((const float4*)src)[i];
    ushort4 o;
    o.x = f2bf(f.x); o.y = f2bf(f.y); o.z = f2bf(f.z); o.w = f2bf(f.w);
    ((ushort4*)dst)[i] = o;
  }
}

// ---------------------------------------------------------------------------
// Generic GEMM: C[M,N] = A[M,K] @ W[N,K]^T + bias.  A and W are bf16.
// One wave computes a 16x64 strip (4 accumulators share one A fragment).
// Block = 256 threads = 8 waves stacked in M. Grid = (M/128, ceil(N/64)).
// K loop is double-buffered: fragments for step i+1 load during WMMAs of i.
// MODE 0: store bf16 row-major
// MODE 1: store f32 row-major, += resid
// MODE 2: store f32 transposed out[b, n, t]          (m = b*1024 + t)
// MODE 3: store bf16 transposed  C[(b*N + n)*T + t]  (V -> Vt layout)
// ---------------------------------------------------------------------------
template <int MODE>
__global__ void k_gemm(const u16* __restrict__ A, const u16* __restrict__ W,
                       const float* __restrict__ bias,
                       const float* __restrict__ resid,
                       void* __restrict__ Cv, int M, int N, int K) {
  const int tid  = threadIdx.x;
  const int wave = tid >> 5;
  const int lane = tid & 31;
  const int half = lane >> 4;     // 0: lanes 0-15, 1: lanes 16-31
  const int l16  = lane & 15;

  const int tile_m  = blockIdx.x * 128 + wave * 16;
  const int tile_n0 = blockIdx.y * 64;
  if (tile_m >= M) return;
  int nsub = (N - tile_n0) >> 4;          // # of 16-wide subtiles (uniform)
  if (nsub > 4) nsub = 4;

  const u16* arow  = A + (size_t)(tile_m + l16) * K;
  const u16* brow0 = W + (size_t)(tile_n0 + l16) * K;

  // A fragment (16x32): lanes 0-15 K{0..7,16..23}, lanes 16-31 K{8..15,24..31}
  auto loadA = [&](AFrag& a, int kb) {
    const u16* ap = arow + kb + half * 8;
    a.q2[0] = *(const uint4*)(ap);
    a.q2[1] = *(const uint4*)(ap + 16);
  };
  // B fragment (32x16): lanes 0-15 K{0..7,16..23}, lanes 16-31 K{8..15,24..31}
  auto loadB = [&](AFrag& b, int ns, int kb) {
    const u16* bp = brow0 + (size_t)ns * 16 * K + kb + half * 8;
    b.q2[0] = *(const uint4*)(bp);
    b.q2[1] = *(const uint4*)(bp + 16);
  };

  CFrag c[4];
#pragma unroll
  for (int ns = 0; ns < 4; ++ns)
#pragma unroll
    for (int r = 0; r < 8; ++r) c[ns].f[r] = 0.f;

  AFrag a0, a1, b0[4], b1[4];
  loadA(a0, 0);
#pragma unroll
  for (int ns = 0; ns < 4; ++ns)
    if (ns < nsub) loadB(b0[ns], ns, 0);

  for (int kb = 0; kb < K; kb += 64) {          // K is a multiple of 64
    // prefetch second half of this 64-chunk
    loadA(a1, kb + 32);
#pragma unroll
    for (int ns = 0; ns < 4; ++ns)
      if (ns < nsub) loadB(b1[ns], ns, kb + 32);
    // compute first half
#pragma unroll
    for (int ns = 0; ns < 4; ++ns)
      if (ns < nsub) c[ns].v = wmma_bf16(a0.v, b0[ns].v, c[ns].v);
    // prefetch first half of next 64-chunk
    if (kb + 64 < K) {
      loadA(a0, kb + 64);
#pragma unroll
      for (int ns = 0; ns < 4; ++ns)
        if (ns < nsub) loadB(b0[ns], ns, kb + 64);
    }
    // compute second half
#pragma unroll
    for (int ns = 0; ns < 4; ++ns)
      if (ns < nsub) c[ns].v = wmma_bf16(a1.v, b1[ns].v, c[ns].v);
  }

#pragma unroll
  for (int ns = 0; ns < 4; ++ns) {
    if (ns < nsub) {
#pragma unroll
      for (int r = 0; r < 8; ++r) {
        const int m = tile_m + r + half * 8;
        const int n = tile_n0 + ns * 16 + l16;
        float val = c[ns].f[r] + bias[n];
        if (MODE == 0) {
          ((u16*)Cv)[(size_t)m * N + n] = f2bf(val);
        } else if (MODE == 1) {
          val += resid[(size_t)m * N + n];
          ((float*)Cv)[(size_t)m * N + n] = val;
        } else if (MODE == 2) {
          const int b = m >> 10, t = m & 1023;
          ((float*)Cv)[((size_t)b * N + n) * TLEN + t] = val;
        } else {  // MODE 3
          const int b = m >> 10, t = m & 1023;
          ((u16*)Cv)[((size_t)b * N + n) * TLEN + t] = f2bf(val);
        }
      }
    }
  }
}

// ---------------------------------------------------------------------------
// Flash attention: one wave per (b, h, 16-query tile). HD=64, T=1024.
// Q,K are bf16 [B,T,D]; V is bf16 transposed [B,D,T]; output bf16 [B,T,D].
// K and V fragments for a key block are fetched up-front so the V loads
// overlap the entire LDS softmax phase before the P@V WMMAs consume them.
// ---------------------------------------------------------------------------
__global__ __launch_bounds__(32)
void k_flash_attn(const u16* __restrict__ Q, const u16* __restrict__ Kb,
                  const u16* __restrict__ Vt, u16* __restrict__ O) {
  __shared__ float sS[16][33];
  __shared__ u16   pS[16][32];
  __shared__ float mS[16], lS[16], aS[16];

  const int lane = threadIdx.x;
  const int half = lane >> 4, l16 = lane & 15;
  const int qt = blockIdx.x & 63;
  const int h  = (blockIdx.x >> 6) & 7;
  const int b  = blockIdx.x >> 9;
  const int qbase = qt * 16;

  // Q fragments for d-chunks 0,1 (each 16x32 over head dim)
  AFrag qa[2];
  {
    const u16* qr = Q + ((size_t)(b * TLEN + qbase + l16)) * DMODEL + h * HEADD;
#pragma unroll
    for (int dc = 0; dc < 2; ++dc) {
      const u16* p0 = qr + dc * 32 + half * 8;
      qa[dc].q2[0] = *(const uint4*)(p0);
      qa[dc].q2[1] = *(const uint4*)(p0 + 16);
    }
  }

  if (lane < 16) { mS[lane] = -3.0e38f; lS[lane] = 0.f; }
  CFrag o[4];
#pragma unroll
  for (int dc = 0; dc < 4; ++dc)
#pragma unroll
    for (int r = 0; r < 8; ++r) o[dc].f[r] = 0.f;
  __syncthreads();

  for (int kb = 0; kb < TLEN; kb += 32) {
    // ---- fetch all K fragments (2 key-halves x 2 d-chunks) ----
    AFrag kf[2][2];
#pragma unroll
    for (int kh = 0; kh < 2; ++kh) {
      const u16* kr =
          Kb + ((size_t)(b * TLEN + kb + kh * 16 + l16)) * DMODEL + h * HEADD;
      kf[kh][0].q2[0] = *(const uint4*)(kr + half * 8);
      kf[kh][0].q2[1] = *(const uint4*)(kr + half * 8 + 16);
      kf[kh][1].q2[0] = *(const uint4*)(kr + 32 + half * 8);
      kf[kh][1].q2[1] = *(const uint4*)(kr + 32 + half * 8 + 16);
    }
    // ---- fetch all V fragments early (consumed after softmax) ----
    AFrag vf[4];
#pragma unroll
    for (int dc = 0; dc < 4; ++dc) {
      const u16* vr =
          Vt + ((size_t)(b * DMODEL + h * HEADD + dc * 16 + l16)) * TLEN +
          kb + half * 8;
      vf[dc].q2[0] = *(const uint4*)(vr);
      vf[dc].q2[1] = *(const uint4*)(vr + 16);
    }

    // ---- S = (Q @ K^T) / 8 for 32 keys ----
#pragma unroll
    for (int kh = 0; kh < 2; ++kh) {
      CFrag s;
#pragma unroll
      for (int r = 0; r < 8; ++r) s.f[r] = 0.f;
      s.v = wmma_bf16(qa[0].v, kf[kh][0].v, s.v);
      s.v = wmma_bf16(qa[1].v, kf[kh][1].v, s.v);
#pragma unroll
      for (int r = 0; r < 8; ++r)
        sS[r + half * 8][kh * 16 + l16] = s.f[r] * 0.125f;
    }
    __syncthreads();

    // ---- online softmax row update (16 rows on lanes 0-15) ----
    if (lane < 16) {
      const int row = lane;
      float rmax = -3.0e38f;
#pragma unroll
      for (int j = 0; j < 32; ++j) rmax = fmaxf(rmax, sS[row][j]);
      const float mold = mS[row];
      const float mnew = fmaxf(mold, rmax);
      const float alpha = __expf(mold - mnew);
      float sum = 0.f;
#pragma unroll
      for (int j = 0; j < 32; ++j) {
        const float p = __expf(sS[row][j] - mnew);
        sum += p;
        pS[row][j] = f2bf(p);
      }
      lS[row] = lS[row] * alpha + sum;
      mS[row] = mnew;
      aS[row] = alpha;
    }
    __syncthreads();

    // ---- rescale accumulators by alpha ----
#pragma unroll
    for (int dc = 0; dc < 4; ++dc)
#pragma unroll
      for (int r = 0; r < 8; ++r) o[dc].f[r] *= aS[r + half * 8];

    // ---- P fragment from LDS (A layout, K-dim = key within block) ----
    AFrag pf;
#pragma unroll
    for (int j = 0; j < 8; ++j) pf.u[j]     = pS[l16][half * 8 + j];
#pragma unroll
    for (int j = 0; j < 8; ++j) pf.u[8 + j] = pS[l16][16 + half * 8 + j];

    // ---- O += P @ V (V fragments already resident) ----
#pragma unroll
    for (int dc = 0; dc < 4; ++dc)
      o[dc].v = wmma_bf16(pf.v, vf[dc].v, o[dc].v);
    __syncthreads();
  }

  // ---- finalize: divide by l, write bf16 out ----
#pragma unroll
  for (int dc = 0; dc < 4; ++dc)
#pragma unroll
    for (int r = 0; r < 8; ++r) {
      const int row = r + half * 8;
      O[((size_t)(b * TLEN + qbase + row)) * DMODEL + h * HEADD + dc * 16 + l16] =
          f2bf(o[dc].f[r] / lS[row]);
    }
}

// ---------------------------------------------------------------------------
// LayerNorm with affine params: Y(bf16) = LN(X) * g + b  (one block per row)
// ---------------------------------------------------------------------------
__global__ void k_ln(const float* __restrict__ X, const float* __restrict__ g,
                     const float* __restrict__ bta, u16* __restrict__ Y) {
  __shared__ float red[256];
  const int row = blockIdx.x, tid = threadIdx.x;
  const float* xr = X + (size_t)row * DMODEL;
  const float x0 = xr[tid], x1 = xr[tid + 256];

  red[tid] = x0 + x1;
  __syncthreads();
  for (int s = 128; s > 0; s >>= 1) {
    if (tid < s) red[tid] += red[tid + s];
    __syncthreads();
  }
  const float mu = red[0] * (1.f / DMODEL);
  __syncthreads();
  const float d0 = x0 - mu, d1 = x1 - mu;
  red[tid] = d0 * d0 + d1 * d1;
  __syncthreads();
  for (int s = 128; s > 0; s >>= 1) {
    if (tid < s) red[tid] += red[tid + s];
    __syncthreads();
  }
  const float rs = rsqrtf(red[0] * (1.f / DMODEL) + EPS);
  u16* yr = Y + (size_t)row * DMODEL;
  yr[tid]       = f2bf(d0 * rs * g[tid]       + bta[tid]);
  yr[tid + 256] = f2bf(d1 * rs * g[tid + 256] + bta[tid + 256]);
}

// ---------------------------------------------------------------------------
// AdaIN apply (in place, f32): X = LN(X) * (1 + sc[b]) + bi[b]
// ---------------------------------------------------------------------------
__global__ void k_adain(float* __restrict__ X, const float* __restrict__ sc,
                        const float* __restrict__ bi) {
  __shared__ float red[256];
  const int row = blockIdx.x, tid = threadIdx.x;
  const int b = row >> 10;
  float* xr = X + (size_t)row * DMODEL;
  const float x0 = xr[tid], x1 = xr[tid + 256];

  red[tid] = x0 + x1;
  __syncthreads();
  for (int s = 128; s > 0; s >>= 1) {
    if (tid < s) red[tid] += red[tid + s];
    __syncthreads();
  }
  const float mu = red[0] * (1.f / DMODEL);
  __syncthreads();
  const float d0 = x0 - mu, d1 = x1 - mu;
  red[tid] = d0 * d0 + d1 * d1;
  __syncthreads();
  for (int s = 128; s > 0; s >>= 1) {
    if (tid < s) red[tid] += red[tid + s];
    __syncthreads();
  }
  const float rs = rsqrtf(red[0] * (1.f / DMODEL) + EPS);
  const float* sb = sc + b * DMODEL;
  const float* bb = bi + b * DMODEL;
  xr[tid]       = d0 * rs * (1.f + sb[tid])       + bb[tid];
  xr[tid + 256] = d1 * rs * (1.f + sb[tid + 256]) + bb[tid + 256];
}

// ---------------------------------------------------------------------------
// SwiGLU (bf16 in / bf16 out, in place into g1): g1 = silu(g1) * g2
// ---------------------------------------------------------------------------
__global__ void k_swiglu(u16* __restrict__ g1, const u16* __restrict__ g2,
                         size_t n) {
  const size_t idx = (size_t)blockIdx.x * 256 + threadIdx.x;
  if (idx < n) {
    const float a = bf2f(g1[idx]);
    g1[idx] = f2bf((a / (1.f + __expf(-a))) * bf2f(g2[idx]));
  }
}

// ---------------------------------------------------------------------------
// Input projection: x[b,t,d] = sum_m x_t[b,m,t] * in_w[d,m] + in_b[d]  (K=80)
// ---------------------------------------------------------------------------
__global__ void k_inproj(const float* __restrict__ xt,
                         const float* __restrict__ w,
                         const float* __restrict__ bias,
                         float* __restrict__ X) {
  const int idx = blockIdx.x * 256 + threadIdx.x;   // b*T*D + t*D + d
  const int d = idx & 511;
  const int t = (idx >> 9) & 1023;
  const int b = idx >> 19;
  const float* xtp = xt + (size_t)b * MEL * TLEN + t;
  const float* wr  = w + d * MEL;
  float acc = bias[d];
#pragma unroll 8
  for (int m = 0; m < MEL; ++m) acc += xtp[m * TLEN] * wr[m];
  X[idx] = acc;
}

// ---------------------------------------------------------------------------
// Time embedding: sinusoidal -> 2-layer MLP (relu).  One block, 256 threads.
// ---------------------------------------------------------------------------
__global__ void k_temb(const float* __restrict__ t,
                       const float* __restrict__ w1, const float* __restrict__ b1,
                       const float* __restrict__ w2, const float* __restrict__ b2,
                       float* __restrict__ te) {
  __shared__ float emb[BATCH][TEMB];
  __shared__ float h1[BATCH][TEMB];
  const int tid = threadIdx.x;
  const int j = (tid < 128) ? tid : tid - 128;
  const float freq = __expf(-(float)j * (9.210340372f / 128.f)); // ln(1e4)/128
  for (int b = 0; b < BATCH; ++b) {
    const float a = t[b] * freq;
    emb[b][tid] = (tid < 128) ? __sinf(a) : __cosf(a);
  }
  __syncthreads();
  for (int b = 0; b < BATCH; ++b) {
    float acc = b1[tid];
    const float* wr = w1 + tid * TEMB;
    for (int k = 0; k < TEMB; ++k) acc += emb[b][k] * wr[k];
    h1[b][tid] = fmaxf(acc, 0.f);
  }
  __syncthreads();
  for (int b = 0; b < BATCH; ++b) {
    float acc = b2[tid];
    const float* wr = w2 + tid * TEMB;
    for (int k = 0; k < TEMB; ++k) acc += h1[b][k] * wr[k];
    te[b * TEMB + tid] = acc;
  }
}

// ---------------------------------------------------------------------------
// AdaIN scale/bias vectors for all layers & batches: [L][B][D] each.
// ---------------------------------------------------------------------------
__global__ void k_adain_params(const float* __restrict__ te,
                               const float* __restrict__ spk,
                               const float* __restrict__ ta_sw,
                               const float* __restrict__ ta_sb,
                               const float* __restrict__ ta_bw,
                               const float* __restrict__ ta_bb,
                               const float* __restrict__ sa_sw,
                               const float* __restrict__ sa_sb,
                               const float* __restrict__ sa_bw,
                               const float* __restrict__ sa_bb,
                               float* __restrict__ sct, float* __restrict__ bit,
                               float* __restrict__ scs, float* __restrict__ bis) {
  const int idx = blockIdx.x * 256 + threadIdx.x;   // l*(4*512) + b*512 + d
  const int d = idx & 511;
  const int b = (idx >> 9) & 3;
  const int l = idx >> 11;
  const float* tv = te + b * TEMB;
  const float* sv = spk + b * SPK;
  const size_t wd = (size_t)(l * DMODEL + d);

  float s1 = ta_sb[wd], b1 = ta_bb[wd];
  const float* w1 = ta_sw + wd * TEMB;
  const float* w2 = ta_bw + wd * TEMB;
  for (int k = 0; k < TEMB; ++k) { s1 += tv[k] * w1[k]; b1 += tv[k] * w2[k]; }

  float s2 = sa_sb[wd], b2 = sa_bb[wd];
  const float* w3 = sa_sw + wd * SPK;
  const float* w4 = sa_bw + wd * SPK;
  for (int k = 0; k < SPK; ++k) { s2 += sv[k] * w3[k]; b2 += sv[k] * w4[k]; }

  sct[idx] = s1; bit[idx] = b1; scs[idx] = s2; bis[idx] = b2;
}

// ---------------------------------------------------------------------------
// Host-side orchestration
// ---------------------------------------------------------------------------
extern "C" void kernel_launch(void* const* d_in, const int* in_sizes, int n_in,
                              void* d_out, int out_size, void* d_ws,
                              size_t ws_size, hipStream_t stream) {
  (void)in_sizes; (void)n_in; (void)out_size; (void)ws_size;

  const float* xt    = (const float*)d_in[0];
  const float* tt    = (const float*)d_in[1];
  const float* spk   = (const float*)d_in[2];
  const float* in_w  = (const float*)d_in[3];
  const float* in_b  = (const float*)d_in[4];
  const float* tm_w1 = (const float*)d_in[5];
  const float* tm_b1 = (const float*)d_in[6];
  const float* tm_w2 = (const float*)d_in[7];
  const float* tm_b2 = (const float*)d_in[8];
  const float* norm_g = (const float*)d_in[9];
  const float* norm_b = (const float*)d_in[10];
  const float* q_w = (const float*)d_in[11];
  const float* q_b = (const float*)d_in[12];
  const float* k_w = (const float*)d_in[13];
  const float* k_b = (const float*)d_in[14];
  const float* v_w = (const float*)d_in[15];
  const float* v_b = (const float*)d_in[16];
  const float* o_w = (const float*)d_in[17];
  const float* o_b = (const float*)d_in[18];
  const float* ta_sw = (const float*)d_in[19];
  const float* ta_sb = (const float*)d_in[20];
  const float* ta_bw = (const float*)d_in[21];
  const float* ta_bb = (const float*)d_in[22];
  const float* sa_sw = (const float*)d_in[23];
  const float* sa_sb = (const float*)d_in[24];
  const float* sa_bw = (const float*)d_in[25];
  const float* sa_bb = (const float*)d_in[26];
  const float* f_w1 = (const float*)d_in[27];
  const float* f_b1 = (const float*)d_in[28];
  const float* f_w2 = (const float*)d_in[29];
  const float* f_b2 = (const float*)d_in[30];
  const float* f_w3 = (const float*)d_in[31];
  const float* f_b3 = (const float*)d_in[32];
  const float* fn_g = (const float*)d_in[33];
  const float* fn_b = (const float*)d_in[34];
  const float* out_w = (const float*)d_in[35];
  const float* out_b = (const float*)d_in[36];

  const size_t S  = (size_t)NROWS * DMODEL;     // 2,097,152
  const size_t DD = (size_t)DMODEL * DMODEL;    //   262,144
  const size_t FD = (size_t)FFN * DMODEL;       // 1,048,576

  float* ws = (float*)d_ws;
  float* x   = ws;                               // [4096,512] f32
  u16*  hb   = (u16*)(ws + S);                   // LN output, bf16
  u16*  qb   = hb + S;                           // Q bf16 [B,T,D]
  u16*  kbuf = qb + S;                           // K bf16 [B,T,D]
  u16*  vt   = kbuf + S;                         // V bf16 transposed [B,D,T]
  u16*  ab   = vt + S;                           // attn out bf16 [B,T,D]
  u16*  g1b  = ab + S;                           // [4096,2048] bf16
  u16*  g2b  = g1b + 4 * S;                      // [4096,2048] bf16
  u16*  wq   = g2b + 4 * S;                      // per-layer bf16 weights
  u16*  wk   = wq + DD;
  u16*  wv   = wk + DD;
  u16*  wo   = wv + DD;
  u16*  w1b  = wo + DD;
  u16*  w2b  = w1b + FD;
  u16*  w3b  = w2b + FD;
  u16*  wout = w3b + FD;                         // out_w bf16 (80*512)
  float* te  = (float*)(wout + (size_t)MEL * DMODEL);
  float* sct = te + BATCH * TEMB;
  float* bit = sct + L_LAYERS * BATCH * DMODEL;
  float* scs = bit + L_LAYERS * BATCH * DMODEL;
  float* bis = scs + L_LAYERS * BATCH * DMODEL;

  k_temb<<<1, 256, 0, stream>>>(tt, tm_w1, tm_b1, tm_w2, tm_b2, te);
  k_adain_params<<<(L_LAYERS * BATCH * DMODEL) / 256, 256, 0, stream>>>(
      te, spk, ta_sw, ta_sb, ta_bw, ta_bb, sa_sw, sa_sb, sa_bw, sa_bb,
      sct, bit, scs, bis);
  k_inproj<<<(NROWS * DMODEL) / 256, 256, 0, stream>>>(xt, in_w, in_b, x);
  k_cvt<<<(int)(MEL * DMODEL / 4 + 255) / 256, 256, 0, stream>>>(
      out_w, wout, (int)(MEL * DMODEL / 4));

  const dim3 gD(NROWS / 128, DMODEL / 64);   // 32 x 8
  const dim3 gF(NROWS / 128, FFN / 64);      // 32 x 32
  const dim3 gO(NROWS / 128, 2);             // N = 80 (tail masked per wave)
  const int cDD = (int)(DD / 4), cFD = (int)(FD / 4);

  for (int l = 0; l < L_LAYERS; ++l) {
    const size_t oDD = (size_t)l * DD;
    const size_t oFD = (size_t)l * FD;

    // one-time bf16 conversion of this layer's weights
    k_cvt<<<cDD / 256, 256, 0, stream>>>(q_w + oDD, wq, cDD);
    k_cvt<<<cDD / 256, 256, 0, stream>>>(k_w + oDD, wk, cDD);
    k_cvt<<<cDD / 256, 256, 0, stream>>>(v_w + oDD, wv, cDD);
    k_cvt<<<cDD / 256, 256, 0, stream>>>(o_w + oDD, wo, cDD);
    k_cvt<<<cFD / 256, 256, 0, stream>>>(f_w1 + oFD, w1b, cFD);
    k_cvt<<<cFD / 256, 256, 0, stream>>>(f_w2 + oFD, w2b, cFD);
    k_cvt<<<cFD / 256, 256, 0, stream>>>(f_w3 + oFD, w3b, cFD);

    k_ln<<<NROWS, 256, 0, stream>>>(x, norm_g + l * DMODEL,
                                    norm_b + l * DMODEL, hb);
    k_gemm<0><<<gD, 256, 0, stream>>>(hb, wq, q_b + l * DMODEL, nullptr,
                                      qb, NROWS, DMODEL, DMODEL);
    k_gemm<0><<<gD, 256, 0, stream>>>(hb, wk, k_b + l * DMODEL, nullptr,
                                      kbuf, NROWS, DMODEL, DMODEL);
    k_gemm<3><<<gD, 256, 0, stream>>>(hb, wv, v_b + l * DMODEL, nullptr,
                                      vt, NROWS, DMODEL, DMODEL);
    k_flash_attn<<<BATCH * NHEAD * (TLEN / 16), 32, 0, stream>>>(qb, kbuf,
                                                                 vt, ab);
    k_gemm<1><<<gD, 256, 0, stream>>>(ab, wo, o_b + l * DMODEL, x,
                                      x, NROWS, DMODEL, DMODEL);
    k_adain<<<NROWS, 256, 0, stream>>>(x, sct + l * BATCH * DMODEL,
                                       bit + l * BATCH * DMODEL);
    k_adain<<<NROWS, 256, 0, stream>>>(x, scs + l * BATCH * DMODEL,
                                       bis + l * BATCH * DMODEL);
    k_ln<<<NROWS, 256, 0, stream>>>(x, fn_g + l * DMODEL,
                                    fn_b + l * DMODEL, hb);
    k_gemm<0><<<gF, 256, 0, stream>>>(hb, w1b, f_b1 + l * FFN, nullptr,
                                      g1b, NROWS, FFN, DMODEL);
    k_gemm<0><<<gF, 256, 0, stream>>>(hb, w2b, f_b2 + l * FFN, nullptr,
                                      g2b, NROWS, FFN, DMODEL);
    k_swiglu<<<(NROWS * FFN) / 256, 256, 0, stream>>>(g1b, g2b,
                                                      (size_t)NROWS * FFN);
    k_gemm<1><<<gD, 256, 0, stream>>>(g1b, w3b, f_b3 + l * DMODEL, x,
                                      x, NROWS, DMODEL, FFN);
  }

  // final projection: convert residual stream to bf16, then GEMM (N = 80)
  k_cvt<<<(int)(S / 4) / 256, 256, 0, stream>>>(x, hb, (int)(S / 4));
  k_gemm<2><<<gO, 256, 0, stream>>>(hb, wout, out_b, nullptr, (float*)d_out,
                                    NROWS, MEL, DMODEL);
}
